// WeightModulatedINR_61641370632322
// MI455X (gfx1250) — matrix operand order repaired and checked by reading, not compiled, not run on hardware
//
#include <hip/hip_runtime.h>

typedef __bf16 bf16;
typedef __attribute__((ext_vector_type(16))) __bf16 bf16x16;
typedef __attribute__((ext_vector_type(8)))  __bf16 bf16x8;
typedef __attribute__((ext_vector_type(8)))  float  v8f;
typedef __attribute__((ext_vector_type(4)))  unsigned v4u;
typedef __attribute__((ext_vector_type(8)))  int      v8i;
typedef __attribute__((ext_vector_type(4)))  int      v4i;

#define BATCH  8
#define NPTS   32768
#define ROWS   (BATCH * NPTS)   // 262144 points
#define HID    256
#define SDIM   512
#define STRIDE 288              // row stride (bf16 elems) for all activation buffers

#if defined(__has_builtin)
#if __has_builtin(__builtin_amdgcn_tensor_load_to_lds) && \
    __has_builtin(__builtin_amdgcn_s_wait_tensorcnt)
#define TDM_OK 1
#endif
#endif

// ---------------------------------------------------------------------------
// Packed B-matrix (weight) layout for v_wmma_f32_16x16x32_bf16:
//   tile (kt,nt) covers k in [kt*32,kt*32+32), n in [nt*16,nt*16+16)
//   lane = (kr/16)*16 + nr   (lanes 0-15: k 0..15, lanes 16-31: k 16..31)
//   per-lane 16 contiguous bf16 (32B), element e -> k = kt*32 + (lane/16)*16 + e
// ---------------------------------------------------------------------------
__device__ __forceinline__ size_t packed_off(int k, int n) {
  const int kt = k >> 5, kr = k & 31;
  const int nt = n >> 4, nr = n & 15;
  const int lane = ((kr >> 4) << 4) | nr;
  const int e = kr & 15;
  return ((size_t)(kt * 16 + nt) * 32 + (size_t)lane) * 16 + (size_t)e;
}

// ---- per-(b,s) inverse L2 norm over channel dim of feat [B,H,S] -----------
__global__ void __launch_bounds__(256) k_invnorm(const float* __restrict__ feat,
                                                 float* __restrict__ invn) {
  const int t = blockIdx.x * 256 + threadIdx.x;     // b*S + s
  if (t >= BATCH * SDIM) return;
  const int b = t / SDIM, s = t % SDIM;
  const float* f = feat + (size_t)b * HID * SDIM + s;
  float acc = 0.0f;
  for (int c = 0; c < HID; ++c) { const float v = f[(size_t)c * SDIM]; acc = fmaf(v, v, acc); }
  invn[t] = 1.0f / fmaxf(sqrtf(acc), 1e-12f);
}

// ---- pack static weight w [HID][Ksrc] f32 -> bf16 B-fragments (Kp=256) ----
__global__ void __launch_bounds__(256) k_pack_w(const float* __restrict__ w,
                                                bf16* __restrict__ wp, int Ksrc) {
  const int t = blockIdx.x * 256 + threadIdx.x;     // over 256*256
  if (t >= 256 * 256) return;
  const int k = t >> 8, n = t & 255;
  const float v = (k < Ksrc) ? w[(size_t)n * Ksrc + k] : 0.0f;
  wp[packed_off(k, n)] = (bf16)v;
}

// ---- per-batch modulation matrix, packed (Kp=288): rows 0..255 = proj+pb,
//      row 256 = mb, rows 257..287 = 0 ------------------------------------
__global__ void __launch_bounds__(256) k_proj(const float* __restrict__ feat,
                                              const float* __restrict__ invn,
                                              const float* __restrict__ pw,
                                              const float* __restrict__ pb,
                                              const float* __restrict__ mb,
                                              bf16* __restrict__ projp) {
  const int b = blockIdx.y;
  const int t = blockIdx.x * 256 + threadIdx.x;     // over 288*256
  if (t >= 288 * 256) return;
  const int k = t >> 8, n = t & 255;
  float v;
  if (k < HID) {
    const float* fr  = feat + ((size_t)b * HID + k) * SDIM;
    const float* iv  = invn + (size_t)b * SDIM;
    const float* pwr = pw + (size_t)n * SDIM;
    float acc = 0.0f;
    for (int s = 0; s < SDIM; ++s) acc = fmaf(fr[s] * iv[s], pwr[s], acc);
    v = acc + pb[n];
  } else if (k == HID) {
    v = mb[n];
  } else {
    v = 0.0f;
  }
  projp[(size_t)b * (288 * 256) + packed_off(k, n)] = (bf16)v;
}

// ---- Fourier features: z0[p, c*40+f] = cos, z0[p, 120 + c*40+f] = sin -----
__global__ void __launch_bounds__(256) k_fourier(const float* __restrict__ coord,
                                                 bf16* __restrict__ out) {
  const size_t p = (size_t)blockIdx.x * 256 + threadIdx.x;
  if (p >= ROWS) return;
  const float c0 = coord[p * 3 + 0], c1 = coord[p * 3 + 1], c2 = coord[p * 3 + 2];
  bf16* row = out + p * STRIDE;
  for (int f = 0; f < 40; ++f) {
    const float fr = exp2f((10.0f / 39.0f) * (float)f);   // 1024^(f/39)
    float s, c;
    __sincosf(c0 * fr, &s, &c); row[f]       = (bf16)c; row[120 + f]       = (bf16)s;
    __sincosf(c1 * fr, &s, &c); row[40 + f]  = (bf16)c; row[160 + f]       = (bf16)s;
    __sincosf(c2 * fr, &s, &c); row[80 + f]  = (bf16)c; row[200 + f]       = (bf16)s;
  }
  #pragma unroll
  for (int k = 240; k < 256; ++k) row[k] = (bf16)0.0f;    // K padding for layer0
}

// ---- generic WMMA layer: out[M,256] = act(in[M,Kp] @ W[Kp,256] + bias) ----
// 256 threads = 8 waves; block owns 128 rows; W staged in LDS (B-fragment order)
// by the Tensor Data Mover (wave 0 issues one tensor_load_to_lds for KT*16KB).
// kt-outer / nt-inner with 16 live accumulators matches the best schedule.
template <int KT, bool HASBIAS, bool RELU, bool APPEND, bool PERBATCH>
__global__ void __launch_bounds__(256, 1)
k_layer(const bf16* __restrict__ in, int in_stride,
        const bf16* __restrict__ wp, size_t w_batch_stride,
        const float* __restrict__ bias,
        bf16* __restrict__ out, int out_stride) {
  extern __shared__ char smem[];
  bf16* wlds = (bf16*)smem;

  const int tid = threadIdx.x;
  const size_t row0 = (size_t)blockIdx.x * 128;
  const bf16* W = wp;
  if (PERBATCH) {
    const int batch = (int)(row0 / NPTS);             // 128-row tiles never straddle batches
    W += (size_t)batch * w_batch_stride;
  }

#if TDM_OK
  // ---- TDM: DMA the whole weight matrix (KT*16 KB) global -> LDS ----------
  // D# describes a 2D tile of 8-byte elements: KT rows x 2048 elems, contiguous.
  if (tid < 32) {                                     // wave 0 issues the DMA
    const unsigned lds_off = (unsigned)(uintptr_t)(void*)wlds;
    const unsigned long long ga = (unsigned long long)(uintptr_t)W;
    const v4u g0 = { 1u,                              // count=1, no gather
                     lds_off,                          // lds_addr (bytes)
                     (unsigned)ga,                     // global_addr[31:0]
                     (unsigned)((ga >> 32) & 0x1FFFFFFull) | (2u << 30) }; // [56:32]|type=2
    const v8i g1 = { (int)(3u << 16),                 // data_size=3 (8B), mask=0
                     (int)(2048u << 16),              // tensor_dim0[15:0]=2048
                     (int)((unsigned)KT << 16),       // tensor_dim0 hi=0, tensor_dim1=KT
                     (int)(2048u << 16),              // tensor_dim1 hi=0, tile_dim0=2048
                     KT,                              // tile_dim1=KT, tile_dim2=0
                     2048,                            // tensor_dim0_stride[31:0]
                     0, 0 };                          // stride hi / dim1_stride
    const v4i gz = { 0, 0, 0, 0 };
#if __clang_major__ >= 23
    const v8i gz8 = { 0, 0, 0, 0, 0, 0, 0, 0 };
    __builtin_amdgcn_tensor_load_to_lds(g0, g1, gz, gz, gz8, 0);
#else
    __builtin_amdgcn_tensor_load_to_lds(g0, g1, gz, gz, 0);
#endif
    __builtin_amdgcn_s_wait_tensorcnt(0);
  }
  __syncthreads();
#else
  { // fallback: cooperative global -> LDS copy
    const uint4* src = (const uint4*)W;
    uint4* dst = (uint4*)wlds;
    #pragma unroll 4
    for (int i = tid; i < KT * 1024; i += 256) dst[i] = src[i];
  }
  __syncthreads();
#endif

  const int lane  = tid & 31;
  const int mrow  = lane & 15;
  const int khalf = lane >> 4;
  const size_t m0 = row0 + (size_t)(tid >> 5) * 16;
  const bf16* arow = in + (m0 + (size_t)mrow) * (size_t)in_stride;

  v8f acc[16];
  #pragma unroll
  for (int nt = 0; nt < 16; ++nt) {
    const v8f z = {0.f, 0.f, 0.f, 0.f, 0.f, 0.f, 0.f, 0.f};
    acc[nt] = z;
  }

  // A-fragment per k-tile (contiguous k order, not ISA table order): lane holds M = lane%16;
  // elems 0..15 -> k = khalf*16 + 0..15 (matches packed B order)
  #pragma unroll
  for (int kt = 0; kt < KT; ++kt) {
    const int c0 = kt * 32 + khalf * 16;
    const bf16x8 lo = *(const bf16x8*)(arow + c0);
    const bf16x8 hi = *(const bf16x8*)(arow + c0 + 8);
    const bf16x16 afrag = __builtin_shufflevector(lo, hi, 0, 1, 2, 3, 4, 5, 6, 7,
                                                          8, 9, 10, 11, 12, 13, 14, 15);
    #pragma unroll
    for (int nt = 0; nt < 16; ++nt) {
      const bf16x16 bfrag =
          *(const bf16x16*)(wlds + (((size_t)(kt * 16 + nt) * 32 + lane) << 4));
      acc[nt] = __builtin_amdgcn_wmma_f32_16x16x32_bf16(
          /*neg_a=*/false, afrag, /*neg_b=*/false, bfrag,
          /*c_mod=*/(short)0, acc[nt], /*reuse_a=*/false, /*reuse_b=*/false);
    }
  }

  #pragma unroll
  for (int nt = 0; nt < 16; ++nt) {
    const int ncol = nt * 16 + mrow;
    const float bv = HASBIAS ? bias[ncol] : 0.0f;
    #pragma unroll
    for (int i = 0; i < 8; ++i) {                     // C/D: m = i + khalf*8, n = lane%16
      float v = acc[nt][i] + bv;
      if (RELU) v = fmaxf(v, 0.0f);
      const size_t r = m0 + (size_t)(khalf * 8 + i);
      out[r * (size_t)out_stride + ncol] = (bf16)v;
    }
  }

  if (APPEND && lane < 16) {                          // ones column + zero pad for mod layer
    bf16* rp = out + (m0 + lane) * (size_t)out_stride + HID;
    rp[0] = (bf16)1.0f;
    #pragma unroll
    for (int j = 1; j < 32; ++j) rp[j] = (bf16)0.0f;
  }
}

// ---- output head: [rows,256] bf16 @ w_out[3,256]^T + b_out -> f32 ---------
__global__ void __launch_bounds__(256) k_head(const bf16* __restrict__ z,
                                              const float* __restrict__ wout,
                                              const float* __restrict__ bout,
                                              float* __restrict__ out) {
  const size_t p = (size_t)blockIdx.x * 256 + threadIdx.x;
  if (p >= ROWS) return;
  const bf16* r = z + p * STRIDE;
  float a0 = bout[0], a1 = bout[1], a2 = bout[2];
  for (int k = 0; k < HID; ++k) {
    const float v = (float)r[k];
    a0 = fmaf(v, wout[k], a0);
    a1 = fmaf(v, wout[HID + k], a1);
    a2 = fmaf(v, wout[2 * HID + k], a2);
  }
  out[p * 3 + 0] = a0; out[p * 3 + 1] = a1; out[p * 3 + 2] = a2;
}

// ---------------------------------------------------------------------------
extern "C" void kernel_launch(void* const* d_in, const int* in_sizes, int n_in,
                              void* d_out, int out_size, void* d_ws, size_t ws_size,
                              hipStream_t stream) {
  (void)in_sizes; (void)n_in; (void)out_size; (void)ws_size;
  const float* coord = (const float*)d_in[0];
  const float* feat0 = (const float*)d_in[1];
  const float* feat1 = (const float*)d_in[2];
  const float* w0    = (const float*)d_in[3];
  const float* b0    = (const float*)d_in[4];
  const float* pw0   = (const float*)d_in[5];
  const float* pb0   = (const float*)d_in[6];
  const float* mb0   = (const float*)d_in[7];
  const float* w2    = (const float*)d_in[8];
  const float* b2    = (const float*)d_in[9];
  const float* pw1   = (const float*)d_in[10];
  const float* pb1   = (const float*)d_in[11];
  const float* mb1   = (const float*)d_in[12];
  const float* w4    = (const float*)d_in[13];
  const float* b4    = (const float*)d_in[14];
  const float* wout  = (const float*)d_in[15];
  const float* bout  = (const float*)d_in[16];

  char* ws = (char*)d_ws;
  size_t off = 0;
  auto alloc = [&](size_t bytes) -> char* {
    char* p = ws + off;
    off += (bytes + 255) & ~(size_t)255;
    return p;
  };
  bf16*  bufA   = (bf16*)alloc((size_t)ROWS * STRIDE * sizeof(bf16));   // ~151 MB
  bf16*  bufB   = (bf16*)alloc((size_t)ROWS * STRIDE * sizeof(bf16));   // ~151 MB
  bf16*  w0p    = (bf16*)alloc((size_t)256 * 256 * sizeof(bf16));
  bf16*  w2p    = (bf16*)alloc((size_t)256 * 256 * sizeof(bf16));
  bf16*  w4p    = (bf16*)alloc((size_t)256 * 256 * sizeof(bf16));
  bf16*  proj0p = (bf16*)alloc((size_t)BATCH * 288 * 256 * sizeof(bf16));
  bf16*  proj1p = (bf16*)alloc((size_t)BATCH * 288 * 256 * sizeof(bf16));
  float* invn0  = (float*)alloc((size_t)BATCH * SDIM * sizeof(float));
  float* invn1  = (float*)alloc((size_t)BATCH * SDIM * sizeof(float));

  // template instantiations used below
  auto* lin_app = k_layer<8, true,  true,  true,  false>;  // layer0 / layer2
  auto* lin_fin = k_layer<8, true,  true,  false, false>;  // layer4
  auto* modlay  = k_layer<9, false, false, false, true >;  // mod layers

  (void)hipFuncSetAttribute((const void*)lin_app,
                            hipFuncAttributeMaxDynamicSharedMemorySize, 8 * 16 * 1024);
  (void)hipFuncSetAttribute((const void*)lin_fin,
                            hipFuncAttributeMaxDynamicSharedMemorySize, 8 * 16 * 1024);
  (void)hipFuncSetAttribute((const void*)modlay,
                            hipFuncAttributeMaxDynamicSharedMemorySize, 9 * 16 * 1024);

  const dim3 blk(256);
  k_invnorm<<<dim3((BATCH * SDIM + 255) / 256), blk, 0, stream>>>(feat0, invn0);
  k_invnorm<<<dim3((BATCH * SDIM + 255) / 256), blk, 0, stream>>>(feat1, invn1);
  k_pack_w<<<dim3(256), blk, 0, stream>>>(w0, w0p, 240);
  k_pack_w<<<dim3(256), blk, 0, stream>>>(w2, w2p, 256);
  k_pack_w<<<dim3(256), blk, 0, stream>>>(w4, w4p, 256);
  k_proj<<<dim3(288, BATCH), blk, 0, stream>>>(feat0, invn0, pw0, pb0, mb0, proj0p);
  k_proj<<<dim3(288, BATCH), blk, 0, stream>>>(feat1, invn1, pw1, pb1, mb1, proj1p);
  k_fourier<<<dim3(ROWS / 256), blk, 0, stream>>>(coord, bufA);

  const int nblk = ROWS / 128;                        // 2048 tiles of 128 points
  const size_t pstride = (size_t)288 * 256;           // per-batch packed proj stride
  // layer0: Kp=256, bias+ReLU, append ones column for mod layer
  lin_app<<<dim3(nblk), blk, 8 * 16 * 1024, stream>>>(bufA, STRIDE, w0p, 0, b0, bufB, STRIDE);
  // mod layer 1: Kp=288, per-batch weights, no bias/act
  modlay<<<dim3(nblk), blk, 9 * 16 * 1024, stream>>>(bufB, STRIDE, proj0p, pstride, nullptr, bufA, STRIDE);
  // layer2
  lin_app<<<dim3(nblk), blk, 8 * 16 * 1024, stream>>>(bufA, STRIDE, w2p, 0, b2, bufB, STRIDE);
  // mod layer 3
  modlay<<<dim3(nblk), blk, 9 * 16 * 1024, stream>>>(bufB, STRIDE, proj1p, pstride, nullptr, bufA, STRIDE);
  // layer4
  lin_fin<<<dim3(nblk), blk, 8 * 16 * 1024, stream>>>(bufA, STRIDE, w4p, 0, b4, bufB, STRIDE);
  // output head
  k_head<<<dim3(ROWS / 256), blk, 0, stream>>>(bufB, wout, bout, (float*)d_out);
}
